// EfficientMultiheadSelfAttention_7327214207574
// MI455X (gfx1250) — matrix operand order repaired and checked by reading, not compiled, not run on hardware
//
#include <hip/hip_runtime.h>

typedef __attribute__((ext_vector_type(16))) _Float16 v16h;
typedef __attribute__((ext_vector_type(8)))  _Float16 v8h;
typedef __attribute__((ext_vector_type(8)))  float    v8f;
typedef __attribute__((ext_vector_type(4)))  float    v4f;
typedef __attribute__((ext_vector_type(4)))  unsigned int v4u;
typedef __attribute__((ext_vector_type(8)))  int      v8i;
typedef __attribute__((ext_vector_type(4)))  int      v4i;

#define B_   4
#define N_   16384
#define C_   128
#define H_   128
#define SRR  4
#define NP   1024        // reduced tokens per batch (32*32)
#define DH   64
#define NH   2
#define CH   512         // K/V chunk length held in LDS
#define EPSL 1e-6f

static __device__ __forceinline__ v8f wmma_f16(v16h a, v16h b, v8f c) {
  // D = A(16x32 f16) * B(32x16 f16) + C(16x16 f32)
  return __builtin_amdgcn_wmma_f32_16x16x32_f16(false, a, false, b, (short)0, c,
                                                false, false);
}

static __device__ __forceinline__ v16h combine8(v8h lo, v8h hi) {
  union { v16h v; v8h h[2]; } u;
  u.h[0] = lo; u.h[1] = hi;
  return u.v;
}

// A-fragment (16x32 f16): lane L -> row L%16; halves 0-7 -> k = hg..hg+7,
// halves 8-15 -> k = 16+hg..16+hg+7, with hg = (L/16)*8.
static __device__ __forceinline__ v16h load_a_f16(const _Float16* rowp, int hg) {
  v8h lo = *(const v8h*)(rowp + hg);
  v8h hi = *(const v8h*)(rowp + 16 + hg);
  return combine8(lo, hi);
}

static __device__ __forceinline__ v16h load_a_f32cvt(const float* rowp, int hg) {
  v4f f0 = *(const v4f*)(rowp + hg);
  v4f f1 = *(const v4f*)(rowp + hg + 4);
  v4f f2 = *(const v4f*)(rowp + 16 + hg);
  v4f f3 = *(const v4f*)(rowp + 16 + hg + 4);
  v16h a;
#pragma unroll
  for (int i = 0; i < 4; ++i) {
    a[i]      = (_Float16)f0[i];
    a[4 + i]  = (_Float16)f1[i];
    a[8 + i]  = (_Float16)f2[i];
    a[12 + i] = (_Float16)f3[i];
  }
  return a;
}

// B-fragment (32x16 f16): lane L -> col L%16; halves 0-7 -> k = hg..hg+7,
// halves 8-15 -> k = 16+hg..16+hg+7, hg = (L/16)*8.
// colp points at the column's contiguous k-major data; koff includes (L/16)*16.
static __device__ __forceinline__ v16h load_b(const _Float16* colp, int koff) {
  v8h lo = *(const v8h*)(colp + koff - ((koff & 16) >> 1));
  v8h hi = *(const v8h*)(colp + koff - ((koff & 16) >> 1) + 16);
  return combine8(lo, hi);
}

// LDS byte offset of a generic pointer into shared memory (LDS addr = addr[31:0]).
static __device__ __forceinline__ unsigned lds_off(const void* p) {
  return (unsigned)(unsigned long long)(uintptr_t)p;
}

// ASYNCcnt-tracked 16B copy global -> LDS (no VGPR round trip).
static __device__ __forceinline__ void async_copy_b128(unsigned lds_byte_off,
                                                       const void* gsrc) {
  asm volatile("global_load_async_to_lds_b128 %0, %1, off"
               :: "v"(lds_byte_off), "v"(gsrc)
               : "memory");
}

static __device__ __forceinline__ void wait_asynccnt0() {
  asm volatile("s_wait_asynccnt 0x0" ::: "memory");
}

// ---------------------------------------------------------------------------
// 1) SR conv (stride==kernel==4, 'SAME' => no pad since 128%4==0) + bias + LN.
//    One 128-thread block per reduced token. Output fp32 xn[4096][128].
// ---------------------------------------------------------------------------
__global__ void conv_ln_kernel(const float* __restrict__ x,
                               const float* __restrict__ wsr,   // [4][4][128][128] HWIO
                               const float* __restrict__ bias,
                               const float* __restrict__ gamma,
                               const float* __restrict__ beta,
                               float* __restrict__ xn) {
  __shared__ float patch[SRR * SRR * C_];   // 2048 floats
  __shared__ float reds[8];
  const int m  = blockIdx.x;                // b*1024 + py*32 + px
  const int b  = m >> 10;
  const int py = (m >> 5) & 31;
  const int px = m & 31;
  const int t  = threadIdx.x;               // 0..127

  for (int idx = t; idx < SRR * SRR * C_; idx += 128) {
    int pos = idx >> 7;                     // i*4+j
    int c   = idx & 127;
    int i   = pos >> 2, j = pos & 3;
    int row = (py * 4 + i) * H_ + (px * 4 + j);
    patch[idx] = x[((size_t)b * N_ + row) * C_ + c];
  }
  __syncthreads();

  float acc = bias[t];
  for (int k = 0; k < SRR * SRR * C_; ++k)
    acc = fmaf(patch[k], wsr[(size_t)k * C_ + t], acc);

  // LayerNorm over 128 channels (4 wave32s)
  float s = acc, s2 = acc * acc;
#pragma unroll
  for (int msk = 1; msk < 32; msk <<= 1) {
    s  += __shfl_xor(s,  msk, 32);
    s2 += __shfl_xor(s2, msk, 32);
  }
  int wid = t >> 5, lane = t & 31;
  if (lane == 0) { reds[wid] = s; reds[4 + wid] = s2; }
  __syncthreads();
  float ts  = reds[0] + reds[1] + reds[2] + reds[3];
  float ts2 = reds[4] + reds[5] + reds[6] + reds[7];
  float mu  = ts * (1.0f / C_);
  float var = ts2 * (1.0f / C_) - mu * mu;
  float y = (acc - mu) * rsqrtf(var + EPSL) * gamma[t] + beta[t];
  xn[(size_t)m * C_ + t] = y;
}

// ---------------------------------------------------------------------------
// 2) WMMA GEMM: Out_f16_headsplit = f16(X_f32) @ W.   M x 128 x 128.
//    Block: 256 threads (8 waves) covering 128 rows; W^T staged in LDS fp16.
//    Out layout: [(b*NH + h)][n][DH]  (c = h*64 + d).
// ---------------------------------------------------------------------------
__global__ void proj_f32_kernel(const float* __restrict__ X,
                                const float* __restrict__ W,   // [128][128], k-major
                                _Float16* __restrict__ Out,
                                int rowsPB) {
  __shared__ _Float16 WT[C_ * C_];          // WT[n*128 + k] = W[k][n]
  const int t = threadIdx.x;
  for (int idx = t; idx < C_ * C_; idx += 256) {
    int n = idx >> 7, k = idx & 127;
    WT[idx] = (_Float16)W[(size_t)k * C_ + n];
  }
  __syncthreads();

  const int w  = t >> 5, L = t & 31;
  const int hg = (L >> 4) * 8;
  const int kb = (L >> 4) * 16;
  const int arow = blockIdx.x * 128 + w * 16 + (L & 15);
  const float* xr = X + (size_t)arow * C_;

  v8f acc[8] = {};
#pragma unroll
  for (int kk = 0; kk < C_; kk += 32) {
    v16h a = load_a_f32cvt(xr + kk, hg);
#pragma unroll
    for (int nt = 0; nt < 8; ++nt) {
      v16h bf = load_b(&WT[(nt * 16 + (L & 15)) * C_], kk + kb);
      acc[nt] = wmma_f16(a, bf, acc[nt]);
    }
  }

  // D layout: lane L -> col n = L%16; VGPR r -> row = r + 8*(L/16)
  const int bb      = (blockIdx.x * 128) / rowsPB;   // block fits inside one b
  const int rowbase = blockIdx.x * 128 + w * 16 + 8 * (L >> 4) - bb * rowsPB;
#pragma unroll
  for (int nt = 0; nt < 8; ++nt) {
    int c = nt * 16 + (L & 15);
    int h = c >> 6, d = c & 63;
    _Float16* dst = Out + (((size_t)bb * NH + h) * rowsPB) * DH + d;
#pragma unroll
    for (int r = 0; r < 8; ++r)
      dst[(size_t)(rowbase + r) * DH] = (_Float16)acc[nt][r];
  }
}

// ---------------------------------------------------------------------------
// 3) Fused flash attention per (b,h). Block = 256 thr (8 waves x 16 q-rows).
//    K chunk staged by the Tensor Data Mover; V chunk staged row-major by
//    ASYNC global->LDS copies, then LDS->LDS transposed; online softmax.
// ---------------------------------------------------------------------------
__global__ void attn_kernel(const _Float16* __restrict__ Qh,
                            const _Float16* __restrict__ Kh,
                            const _Float16* __restrict__ Vh,
                            _Float16* __restrict__ AO) {
  extern __shared__ _Float16 smem[];
  _Float16* Ksh    = smem;                   // [CH][DH] f16 (row-major)
  _Float16* Vstage = smem + CH * DH;         // [CH][DH] f16 (row-major staging)
  _Float16* Vt     = smem + 2 * CH * DH;     // [DH][CH] f16 (transposed)
  const int t  = threadIdx.x;
  const int w  = t >> 5, L = t & 31;
  _Float16* pb = smem + 3 * CH * DH + w * (16 * 32);   // per-wave P tile [16][32]

  const int bh = blockIdx.y;
  const int q0 = blockIdx.x * 128 + w * 16;
  const int hg = (L >> 4) * 8;
  const int kb = (L >> 4) * 16;

  const _Float16* qrow = Qh + ((size_t)bh * N_ + q0 + (L & 15)) * DH;
  v16h aq0 = load_a_f16(qrow,      hg);     // dh 0..31
  v16h aq1 = load_a_f16(qrow + 32, hg);     // dh 32..63

  v8f o[4] = {};
  float rm[8], rl[8];
#pragma unroll
  for (int r = 0; r < 8; ++r) { rm[r] = -1e30f; rl[r] = 0.0f; }
  const float scale = 0.125f;               // dh^-0.5

  const unsigned kofs = lds_off(Ksh);
  const unsigned vofs = lds_off(Vstage);

  for (int ck = 0; ck < NP / CH; ++ck) {
    __syncthreads();                        // previous chunk fully consumed

    const _Float16* Ksrc = Kh + ((size_t)bh * NP + ck * CH) * DH;
    const _Float16* Vsrc = Vh + ((size_t)bh * NP + ck * CH) * DH;

    // (a) TDM: one wave DMAs the whole contiguous 64KB K chunk into LDS.
    if (w == 0) {
      const unsigned elems = (CH * DH * (unsigned)sizeof(_Float16)) / 8;  // 8B units
      unsigned long long ga = (unsigned long long)(uintptr_t)Ksrc;
      v4u g0 = { 1u,                                   // count=1, no gather
                 kofs,                                  // lds_addr
                 (unsigned)ga,                          // global_addr[31:0]
                 (unsigned)((ga >> 32) & 0x1FFFFFFu) | (2u << 30) };  // addr[56:32]|type=2
      v8i g1 = { (int)(3u << 16),                       // data_size = 8B
                 (int)((elems & 0xFFFFu) << 16),        // tensor_dim0[15:0]
                 (int)((elems >> 16) | (1u << 16)),     // tensor_dim0[31:16]|tensor_dim1=1
                 (int)((elems & 0xFFFFu) << 16),        // tile_dim0
                 0,                                     // tile_dim1/2 unused
                 (int)elems,                            // tensor_dim0_stride[31:0]
                 0, 0 };
      v4i gz4 = { 0, 0, 0, 0 };
      v8i gz8 = { 0, 0, 0, 0, 0, 0, 0, 0 };
      __builtin_amdgcn_tensor_load_to_lds(g0, g1, gz4, gz4, gz8, 0);
    }

    // (b) ASYNC: V chunk row-major into staging (no VGPR round trip).
    for (int idx = t * 8; idx < CH * DH; idx += 256 * 8)
      async_copy_b128(vofs + (unsigned)idx * 2u, Vsrc + idx);

    wait_asynccnt0();
    if (w == 0) __builtin_amdgcn_s_wait_tensorcnt(0);
    __syncthreads();                        // Ksh + Vstage visible to all waves

    // (c) LDS->LDS transpose Vstage[key][d] -> Vt[d][key]
    for (int idx = t * 8; idx < CH * DH; idx += 256 * 8) {
      int key = idx >> 6;                   // DH == 64
      int d   = idx & 63;
      v8h vv = *(const v8h*)(Vstage + idx);
#pragma unroll
      for (int j = 0; j < 8; ++j) Vt[(d + j) * CH + key] = vv[j];
    }
    __syncthreads();

    for (int c0 = 0; c0 < CH; c0 += 32) {
      // S = Q K^T : two 16x16 key tiles, K-dim = 64 in two 32-steps
      v8f s0 = {}, s1 = {};
      const _Float16* k0p = Ksh + (c0 +      (L & 15)) * DH;
      const _Float16* k1p = Ksh + (c0 + 16 + (L & 15)) * DH;
      s0 = wmma_f16(aq0, load_b(k0p,      kb), s0);
      s0 = wmma_f16(aq1, load_b(k0p + 32, kb), s0);
      s1 = wmma_f16(aq0, load_b(k1p,      kb), s1);
      s1 = wmma_f16(aq1, load_b(k1p + 32, kb), s1);

      float p0[8], p1[8], corr[8];
#pragma unroll
      for (int r = 0; r < 8; ++r) {
        float a0 = s0[r] * scale, a1 = s1[r] * scale;
        float mx = fmaxf(a0, a1);
#pragma unroll
        for (int msk = 1; msk < 16; msk <<= 1)
          mx = fmaxf(mx, __shfl_xor(mx, msk, 32));      // row max (16-lane group)
        float nm = fmaxf(rm[r], mx);
        corr[r]  = __expf(rm[r] - nm);
        rm[r]    = nm;
        p0[r]    = __expf(a0 - nm);
        p1[r]    = __expf(a1 - nm);
        float ps = p0[r] + p1[r];
#pragma unroll
        for (int msk = 1; msk < 16; msk <<= 1)
          ps += __shfl_xor(ps, msk, 32);                // row sum
        rl[r] = rl[r] * corr[r] + ps;
      }
#pragma unroll
      for (int dt = 0; dt < 4; ++dt)
#pragma unroll
        for (int r = 0; r < 8; ++r) o[dt][r] *= corr[r];

      // reshape P (accumulator layout) -> A-fragment layout via per-wave LDS
#pragma unroll
      for (int r = 0; r < 8; ++r) {
        int prow = r + 8 * (L >> 4);
        pb[prow * 32 +      (L & 15)] = (_Float16)p0[r];
        pb[prow * 32 + 16 + (L & 15)] = (_Float16)p1[r];
      }
      v16h ap = load_a_f16(pb + (L & 15) * 32, hg);

      // O += P * V  (4 dh tiles of 16)
#pragma unroll
      for (int dt = 0; dt < 4; ++dt) {
        const _Float16* vp = Vt + (dt * 16 + (L & 15)) * CH + c0;
        o[dt] = wmma_f16(ap, load_b(vp, kb), o[dt]);
      }
    }
  }

  // epilogue: normalize, write fp16 [b][n][h*64+d]
  const int b  = bh >> 1, h = bh & 1;
  const int qb = q0 + 8 * (L >> 4);
#pragma unroll
  for (int r = 0; r < 8; ++r) {
    float inv = 1.0f / rl[r];
    size_t rowoff = ((size_t)b * N_ + qb + r) * C_ + h * DH;
#pragma unroll
    for (int dt = 0; dt < 4; ++dt)
      AO[rowoff + dt * 16 + (L & 15)] = (_Float16)(o[dt][r] * inv);
  }
}

// ---------------------------------------------------------------------------
// 4) Final projection: d_out_f32 = AO_f16 @ Wproj.  65536 x 128 x 128.
// ---------------------------------------------------------------------------
__global__ void proj_f16_kernel(const _Float16* __restrict__ X,
                                const float* __restrict__ W,
                                float* __restrict__ Out) {
  __shared__ _Float16 WT[C_ * C_];
  const int t = threadIdx.x;
  for (int idx = t; idx < C_ * C_; idx += 256) {
    int n = idx >> 7, k = idx & 127;
    WT[idx] = (_Float16)W[(size_t)k * C_ + n];
  }
  __syncthreads();

  const int w  = t >> 5, L = t & 31;
  const int hg = (L >> 4) * 8;
  const int kb = (L >> 4) * 16;
  const int arow = blockIdx.x * 128 + w * 16 + (L & 15);
  const _Float16* xr = X + (size_t)arow * C_;

  v8f acc[8] = {};
#pragma unroll
  for (int kk = 0; kk < C_; kk += 32) {
    v16h a = load_a_f16(xr + kk, hg);
#pragma unroll
    for (int nt = 0; nt < 8; ++nt) {
      v16h bf = load_b(&WT[(nt * 16 + (L & 15)) * C_], kk + kb);
      acc[nt] = wmma_f16(a, bf, acc[nt]);
    }
  }

  const int rowbase = blockIdx.x * 128 + w * 16 + 8 * (L >> 4);
#pragma unroll
  for (int nt = 0; nt < 8; ++nt) {
    int c = nt * 16 + (L & 15);
#pragma unroll
    for (int r = 0; r < 8; ++r)
      Out[(size_t)(rowbase + r) * C_ + c] = acc[nt][r];
  }
}

// ---------------------------------------------------------------------------
extern "C" void kernel_launch(void* const* d_in, const int* in_sizes, int n_in,
                              void* d_out, int out_size, void* d_ws, size_t ws_size,
                              hipStream_t stream) {
  (void)in_sizes; (void)n_in; (void)out_size; (void)ws_size;
  const float* x     = (const float*)d_in[0];
  const float* Wq    = (const float*)d_in[1];
  const float* Wk    = (const float*)d_in[2];
  const float* Wv    = (const float*)d_in[3];
  const float* Wproj = (const float*)d_in[4];
  const float* wsr   = (const float*)d_in[5];
  const float* bsr   = (const float*)d_in[6];
  const float* gamma = (const float*)d_in[7];
  const float* beta  = (const float*)d_in[8];

  char* ws = (char*)d_ws;
  _Float16* Qh = (_Float16*)ws;  ws += (size_t)B_ * NH * N_ * DH * sizeof(_Float16);
  _Float16* Kh = (_Float16*)ws;  ws += (size_t)B_ * NH * NP * DH * sizeof(_Float16);
  _Float16* Vh = (_Float16*)ws;  ws += (size_t)B_ * NH * NP * DH * sizeof(_Float16);
  _Float16* AO = (_Float16*)ws;  ws += (size_t)B_ * N_ * C_ * sizeof(_Float16);
  float*    xn = (float*)ws;     ws += (size_t)B_ * NP * C_ * sizeof(float);

  conv_ln_kernel<<<dim3(B_ * NP), dim3(128), 0, stream>>>(x, wsr, bsr, gamma, beta, xn);
  proj_f32_kernel<<<dim3((B_ * N_) / 128), dim3(256), 0, stream>>>(x,  Wq, Qh, N_);
  proj_f32_kernel<<<dim3((B_ * NP) / 128), dim3(256), 0, stream>>>(xn, Wk, Kh, NP);
  proj_f32_kernel<<<dim3((B_ * NP) / 128), dim3(256), 0, stream>>>(xn, Wv, Vh, NP);

  size_t lds = (size_t)(3 * CH * DH + 8 * 16 * 32) * sizeof(_Float16);   // 204800 B
  attn_kernel<<<dim3(N_ / 128, B_ * NH), dim3(256), lds, stream>>>(Qh, Kh, Vh, AO);

  proj_f16_kernel<<<dim3((B_ * N_) / 128), dim3(256), 0, stream>>>(AO, Wproj, (float*)d_out);
}
